// CrossAttention_40329742910052
// MI455X (gfx1250) — matrix operand order repaired and checked by reading, not compiled, not run on hardware
//
#include <hip/hip_runtime.h>

typedef __attribute__((ext_vector_type(16))) __bf16 v16bf;
typedef __attribute__((ext_vector_type(8)))  float  v8f;

#define BATCH  4
#define C_DIM  256
#define N_PIX  2304
#define NH     4
#define HD     64
#define SCALE  0.125f   // 1/sqrt(64)

// ---------------------------------------------------------------------------
// WMMA fragment helpers (CDNA5 16x16x32 bf16, wave32)
// A-matrix: lanes 0-15 -> M=0..15 (K 0-7,16-23), lanes 16-31 -> same M (K 8-15,24-31)
// B-matrix: lanes 0-15 -> N=0..15 (K 0-7,16-23), lanes 16-31 -> same N (K 8-15,24-31)
// C/D:      lane l -> N = l%16, VGPR r -> M = r + 8*(l/16)
// All call sites keep stride_k == 1 so the 2-byte element reads merge into
// b64/b128 loads (K chunks are contiguous per lane).
// ---------------------------------------------------------------------------
template <bool IS_A, typename T>
__device__ __forceinline__ v16bf load_frag(const T* p, int stride_mn, int stride_k, int lane) {
    const int mn = lane & 15;
    const int kg = lane >> 4;
    v16bf f;
#pragma unroll
    for (int i = 0; i < 8; ++i) {
        const int kk = IS_A ? (((i < 4) ? (2 * i) : (16 + 2 * (i - 4))) + 8 * kg)
                            : (((i < 4) ? (2 * i) : (16 + 2 * (i - 4))) + 8 * kg);
        f[2 * i]     = (__bf16)p[mn * stride_mn + kk * stride_k];
        f[2 * i + 1] = (__bf16)p[mn * stride_mn + (kk + 1) * stride_k];
    }
    return f;
}

__device__ __forceinline__ v8f wmma_bf16(v16bf a, v16bf b, v8f c) {
    return __builtin_amdgcn_wmma_f32_16x16x32_bf16(false, a, false, b, (short)0, c,
                                                   false, false);
}

// ---------------------------------------------------------------------------
// 1x1-conv projection. MODE 0: Y[b][m][n] bf16 (channel-major, for V).
// MODE 1: head-transposed YT[(b*NH + m/64)][n][m%64] bf16 (for Q, K).
// grid (N/16, C/64, B), block 128; X tile staged transposed in LDS (k-contig).
// ---------------------------------------------------------------------------
template <int MODE>
__global__ void __launch_bounds__(128)
gemm_proj_bf16(const float* __restrict__ W, const float* __restrict__ bias,
               const float* __restrict__ X, unsigned short* __restrict__ Yus) {
    __bf16* Y = reinterpret_cast<__bf16*>(Yus);
    const int lane = threadIdx.x & 31;
    const int wave = threadIdx.x >> 5;
    const int b  = blockIdx.z;
    const int n0 = blockIdx.x * 16;
    const int m0 = blockIdx.y * 64 + wave * 16;
    const float* Xb = X + (size_t)b * C_DIM * N_PIX;

    __shared__ __align__(16) __bf16 sXT[16][40];   // [n][k] transposed, padded

    const int srow  = threadIdx.x >> 2;            // 0..31 : k row
    const int squad = threadIdx.x & 3;             // 0..3  : 4 n-columns

    v8f acc = {};
#pragma unroll
    for (int kb = 0; kb < C_DIM; kb += 32) {
        __syncthreads();   // previous iteration finished reading sXT
        {
            const float4 val = *reinterpret_cast<const float4*>(
                Xb + (size_t)(kb + srow) * N_PIX + n0 + squad * 4);
            sXT[squad * 4 + 0][srow] = (__bf16)val.x;
            sXT[squad * 4 + 1][srow] = (__bf16)val.y;
            sXT[squad * 4 + 2][srow] = (__bf16)val.z;
            sXT[squad * 4 + 3][srow] = (__bf16)val.w;
        }
        __syncthreads();
        v16bf a  = load_frag<true >(W + m0 * C_DIM + kb, C_DIM, 1, lane);  // W row-major
        v16bf bm = load_frag<false>(&sXT[0][0], 40, 1, lane);              // k contiguous
        acc = wmma_bf16(a, bm, acc);
    }
    const int hi = lane >> 4, ni = lane & 15;
    if (MODE == 0) {
        __bf16* Yb = Y + (size_t)b * C_DIM * N_PIX;
#pragma unroll
        for (int r = 0; r < 8; ++r) {
            const int m = m0 + r + 8 * hi;
            Yb[(size_t)m * N_PIX + n0 + ni] = (__bf16)(acc[r] + bias[m]);
        }
    } else {
        const int h = m0 >> 6;
        const int dbase = (m0 & 63) + 8 * hi;      // 8 consecutive d per lane
        __align__(16) __bf16 tmp[8];
#pragma unroll
        for (int r = 0; r < 8; ++r) tmp[r] = (__bf16)(acc[r] + bias[m0 + r + 8 * hi]);
        __bf16* dst = Y + ((size_t)(b * NH + h) * N_PIX + n0 + ni) * HD + dbase;
        *reinterpret_cast<uint4*>(dst) = *reinterpret_cast<const uint4*>(tmp);
    }
}

// ---------------------------------------------------------------------------
// Fused flash attention per (b, h, 64-query tile); block = 4 waves,
// wave owns 16 query rows.
//   qT/kT: [(b*NH+h)][n][d] bf16 (d contiguous)   v: [b][h*64+d][n] bf16
//   output aoT: [b][n][c] bf16 (c contiguous)
// ---------------------------------------------------------------------------
__global__ void __launch_bounds__(128)
attn_fused(const unsigned short* __restrict__ qus, const unsigned short* __restrict__ kus,
           const unsigned short* __restrict__ vus, unsigned short* __restrict__ ous) {
    const __bf16* qT = reinterpret_cast<const __bf16*>(qus);
    const __bf16* kT = reinterpret_cast<const __bf16*>(kus);
    const __bf16* v  = reinterpret_cast<const __bf16*>(vus);
    __bf16*      aoT = reinterpret_cast<__bf16*>(ous);

    const int lane = threadIdx.x & 31;
    const int wave = threadIdx.x >> 5;
    const int h = blockIdx.y, b = blockIdx.z;
    const size_t tbase = (size_t)(b * NH + h) * N_PIX * HD;      // qT/kT base
    const size_t vbase = ((size_t)b * C_DIM + h * HD) * N_PIX;   // v base
    const int i0 = blockIdx.x * 64 + wave * 16;                  // wave's query rows

    __shared__ __align__(16) __bf16 sKT[64][72];      // K tile: [j][d], padded
    __shared__ __align__(16) __bf16 sV[64][72];       // V tile: [d][j], padded
    __shared__ float  sS[4][16][66];                  // fp32 scores (padded)
    __shared__ __bf16 sP[4][16][72];                  // exp(S - max), padded
    __shared__ float  sMax[4][16], sSum[4][16], sCorr[4][16];

    // Q^T A-fragments (once): A[m=i_local][kk=d] = qT[(i0+m)*64 + kk], contiguous
    v16bf aq0 = load_frag<true>(qT + tbase + (size_t)i0 * HD,      HD, 1, lane);
    v16bf aq1 = load_frag<true>(qT + tbase + (size_t)i0 * HD + 32, HD, 1, lane);

    v8f ot[4] = {};                                   // O tile: 64(d) x 16(i)
    if (lane < 16) { sMax[wave][lane] = -1e30f; sSum[wave][lane] = 0.0f; }

    const int hi = lane >> 4, ni = lane & 15;

    for (int j0 = 0; j0 < N_PIX; j0 += 64) {
        __syncthreads();   // prior iteration done reading sKT/sV (init visible)
        // ---- stage tiles: kT tile is one contiguous 8KB block ----
        {
            const uint4* gk = reinterpret_cast<const uint4*>(kT + tbase + (size_t)j0 * HD);
            const uint4* gv = reinterpret_cast<const uint4*>(v + vbase + j0);
#pragma unroll
            for (int t = 0; t < 4; ++t) {
                const int e = threadIdx.x + t * 128;   // 0..511
                const int row = e >> 3, c = e & 7;
                *reinterpret_cast<uint4*>(&sKT[row][c * 8]) = gk[e];
                *reinterpret_cast<uint4*>(&sV[row][c * 8])  = gv[row * (N_PIX / 8) + c];
            }
            if (j0 + 64 < N_PIX) {                     // prefetch next tiles
                const int pr = threadIdx.x & 63;
                __builtin_prefetch(kT + tbase + (size_t)(j0 + 64) * HD + pr * HD, 0, 3);
                __builtin_prefetch(v + vbase + (size_t)pr * N_PIX + j0 + 64, 0, 3);
            }
        }
        __syncthreads();

        // ---- S = scale * Q^T K : B[kk=d][n=j] = sKT[j][d], d contiguous ----
#pragma unroll
        for (int nt = 0; nt < 4; ++nt) {
            v8f s = {};
            v16bf bk0 = load_frag<false>(&sKT[nt * 16][0],  72, 1, lane);
            s = wmma_bf16(aq0, bk0, s);
            v16bf bk1 = load_frag<false>(&sKT[nt * 16][32], 72, 1, lane);
            s = wmma_bf16(aq1, bk1, s);
#pragma unroll
            for (int r = 0; r < 8; ++r)
                sS[wave][r + 8 * hi][nt * 16 + ni] = s[r] * SCALE;
        }
        __syncthreads();

        // ---- online softmax, one lane per query row ----
        if (lane < 16) {
            const float mold = sMax[wave][lane];
            float mnew = mold;
            for (int jj = 0; jj < 64; ++jj) mnew = fmaxf(mnew, sS[wave][lane][jj]);
            const float corr = __expf(mold - mnew);
            float lsum = 0.0f;
            for (int jj = 0; jj < 64; ++jj) {
                const float p = __expf(sS[wave][lane][jj] - mnew);
                sP[wave][lane][jj] = (__bf16)p;
                lsum += p;
            }
            sSum[wave][lane]  = sSum[wave][lane] * corr + lsum;
            sMax[wave][lane]  = mnew;
            sCorr[wave][lane] = corr;
        }
        __syncthreads();

        // ---- rescale O accumulator (per query column) ----
        {
            const float c = sCorr[wave][ni];
#pragma unroll
            for (int t = 0; t < 4; ++t)
#pragma unroll
                for (int r = 0; r < 8; ++r) ot[t][r] *= c;
        }

        // ---- O += V * P^T : A[m=d][kk=j] from sV, B[kk=j][n=i] from sP ----
#pragma unroll
        for (int dt = 0; dt < 4; ++dt) {
#pragma unroll
            for (int jc = 0; jc < 64; jc += 32) {
                v16bf av = load_frag<true >(&sV[dt * 16][jc], 72, 1, lane);
                v16bf bp = load_frag<false>(&sP[wave][0][jc], 72, 1, lane);
                ot[dt] = wmma_bf16(av, bp, ot[dt]);
            }
        }
    }
    __syncthreads();

    // ---- normalize + transposed store: aoT[b][i][c], c contiguous ----
    const float rs = 1.0f / sSum[wave][ni];
#pragma unroll
    for (int dt = 0; dt < 4; ++dt) {
        __align__(16) __bf16 tmp[8];
#pragma unroll
        for (int r = 0; r < 8; ++r) tmp[r] = (__bf16)(ot[dt][r] * rs);
        __bf16* dst = aoT + ((size_t)b * N_PIX + i0 + ni) * C_DIM
                          + h * HD + dt * 16 + 8 * hi;
        *reinterpret_cast<uint4*>(dst) = *reinterpret_cast<const uint4*>(tmp);
    }
}

// ---------------------------------------------------------------------------
// Output projection + residual: Y[b,m,n] = sum_c Wo[m,c]*aoT[b,n,c] + bo[m] + x[b,m,n]
// aoT is c-contiguous -> B-fragments load directly from global (no LDS needed).
// ---------------------------------------------------------------------------
__global__ void __launch_bounds__(128)
gemm_oproj(const float* __restrict__ W, const float* __restrict__ bias,
           const unsigned short* __restrict__ Xus, const float* __restrict__ resid,
           float* __restrict__ Y) {
    const __bf16* aoT = reinterpret_cast<const __bf16*>(Xus);
    const int lane = threadIdx.x & 31;
    const int wave = threadIdx.x >> 5;
    const int b  = blockIdx.z;
    const int n0 = blockIdx.x * 16;
    const int m0 = blockIdx.y * 64 + wave * 16;
    const __bf16* Bn = aoT + ((size_t)b * N_PIX + n0) * C_DIM;   // B[kk=c][n] = Bn[n*256+kk]

    v8f acc = {};
#pragma unroll
    for (int kb = 0; kb < C_DIM; kb += 32) {
        v16bf a  = load_frag<true >(W + m0 * C_DIM + kb, C_DIM, 1, lane);
        v16bf bm = load_frag<false>(Bn + kb, C_DIM, 1, lane);
        acc = wmma_bf16(a, bm, acc);
    }
    const int hi = lane >> 4, ni = lane & 15;
    const size_t bofs = (size_t)b * C_DIM * N_PIX;
#pragma unroll
    for (int r = 0; r < 8; ++r) {
        const int m = m0 + r + 8 * hi;
        const size_t idx = bofs + (size_t)m * N_PIX + n0 + ni;
        Y[idx] = acc[r] + bias[m] + resid[idx];
    }
}

// ---------------------------------------------------------------------------
// GroupNorm: 32 groups of 8 channels
// ---------------------------------------------------------------------------
__global__ void __launch_bounds__(256)
gn_stats(const float* __restrict__ y, float* __restrict__ stats) {
    const int g = blockIdx.x, b = blockIdx.y;
    const size_t base = ((size_t)b * C_DIM + g * 8) * N_PIX;
    const int cnt = 8 * N_PIX;
    float s = 0.0f, ss = 0.0f;
    for (int i = threadIdx.x; i < cnt; i += 256) {
        const float vv = y[base + i];
        s += vv; ss += vv * vv;
    }
    __shared__ float rs[256], rss[256];
    rs[threadIdx.x] = s; rss[threadIdx.x] = ss;
    __syncthreads();
    for (int ofs = 128; ofs > 0; ofs >>= 1) {
        if ((int)threadIdx.x < ofs) {
            rs[threadIdx.x]  += rs[threadIdx.x + ofs];
            rss[threadIdx.x] += rss[threadIdx.x + ofs];
        }
        __syncthreads();
    }
    if (threadIdx.x == 0) {
        const float mean = rs[0] / (float)cnt;
        const float var  = rss[0] / (float)cnt - mean * mean;
        stats[(b * 32 + g) * 2 + 0] = mean;
        stats[(b * 32 + g) * 2 + 1] = rsqrtf(var + 1e-5f);
    }
}

__global__ void __launch_bounds__(256)
gn_apply(const float* __restrict__ y, const float* __restrict__ stats,
         const float* __restrict__ gamma, const float* __restrict__ beta,
         float* __restrict__ out) {
    const size_t total = (size_t)BATCH * C_DIM * N_PIX;
    const size_t idx = (size_t)blockIdx.x * 256 + threadIdx.x;
    if (idx >= total) return;
    const int c  = (int)((idx / N_PIX) % C_DIM);
    const int bb = (int)(idx / ((size_t)C_DIM * N_PIX));
    const int g  = c >> 3;
    const float mean = stats[(bb * 32 + g) * 2 + 0];
    const float rinv = stats[(bb * 32 + g) * 2 + 1];
    out[idx] = (y[idx] - mean) * rinv * gamma[c] + beta[c];
}

// ---------------------------------------------------------------------------
extern "C" void kernel_launch(void* const* d_in, const int* in_sizes, int n_in,
                              void* d_out, int out_size, void* d_ws, size_t ws_size,
                              hipStream_t stream) {
    const float* x     = (const float*)d_in[0];
    const float* ctx   = (const float*)d_in[1];
    const float* Wq    = (const float*)d_in[2];
    const float* bq    = (const float*)d_in[3];
    const float* Wk    = (const float*)d_in[4];
    const float* bk    = (const float*)d_in[5];
    const float* Wv    = (const float*)d_in[6];
    const float* bv    = (const float*)d_in[7];
    const float* Wo    = (const float*)d_in[8];
    const float* bo    = (const float*)d_in[9];
    const float* gamma = (const float*)d_in[10];
    const float* beta  = (const float*)d_in[11];
    float* out = (float*)d_out;

    const size_t BCN = (size_t)BATCH * C_DIM * N_PIX;
    char* ws = (char*)d_ws;
    unsigned short* qT  = (unsigned short*)ws; ws += BCN * 2;   // [(b,h)][n][d]
    unsigned short* kT  = (unsigned short*)ws; ws += BCN * 2;   // [(b,h)][n][d]
    unsigned short* vb  = (unsigned short*)ws; ws += BCN * 2;   // [b][c][n]
    unsigned short* aoT = (unsigned short*)ws; ws += BCN * 2;   // [b][n][c]
    float* y     = (float*)ws;                 ws += BCN * 4;
    float* stats = (float*)ws;

    const dim3 gGemm(N_PIX / 16, C_DIM / 64, BATCH);
    gemm_proj_bf16<1><<<gGemm, 128, 0, stream>>>(Wq, bq, x,   qT);
    gemm_proj_bf16<1><<<gGemm, 128, 0, stream>>>(Wk, bk, ctx, kT);
    gemm_proj_bf16<0><<<gGemm, 128, 0, stream>>>(Wv, bv, ctx, vb);

    attn_fused<<<dim3(N_PIX / 64, NH, BATCH), 128, 0, stream>>>(qT, kT, vb, aoT);

    gemm_oproj<<<gGemm, 128, 0, stream>>>(Wo, bo, aoT, x, y);

    gn_stats<<<dim3(32, BATCH), 256, 0, stream>>>(y, stats);
    gn_apply<<<(unsigned)((BCN + 255) / 256), 256, 0, stream>>>(y, stats, gamma, beta, out);
}
